// HierarchicalSoftmaxHead_65730179498404
// MI455X (gfx1250) — matrix operand order repaired and checked by reading, not compiled, not run on hardware
//
#include <hip/hip_runtime.h>
#include <hip/hip_bf16.h>
#include <math.h>

typedef __attribute__((ext_vector_type(16))) _Float16 v16h;
typedef __attribute__((ext_vector_type(8)))  float    v8f;

// Problem constants fixed by the reference
#define DIM      512
#define MAXPATH  18
#define KCHUNKS  (DIM / 32)   // 16 WMMA K-steps of 32

// One wave (32 lanes) per block; each block handles a tile of 16 tokens.
__global__ __launch_bounds__(32)
void hs_path_logits_kernel(const float* __restrict__ hidden,     // (n_tokens, 512) f32
                           const int*   __restrict__ target_ids, // (n_tokens,)
                           const float* __restrict__ node_w,     // (N_internal, 512) f32
                           const int*   __restrict__ path_nodes, // (vocab, 18)
                           const float* __restrict__ path_tgt,   // (vocab, 18)
                           const float* __restrict__ path_msk,   // (vocab, 18)
                           float* __restrict__ ws,               // ws[0]=loss sum, ws[1]=valid count
                           int n_tokens, int vocab)
{
    // B-operand staging: 16 tokens x 512 K in f16, laid out so each lane can
    // reload its v16h chunk with one 32B LDS read per K-chunk.
    __shared__ _Float16 hB[16 * DIM];          // 16 KB
    __shared__ int      s_node[16 * MAXPATH];
    __shared__ float    s_t[16 * MAXPATH];
    __shared__ float    s_m[16 * MAXPATH];

    const int lane  = threadIdx.x;             // 0..31 (wave32)
    const int tbase = blockIdx.x * 16;

    // ---- Stage per-token path metadata (lanes 0..15, one token each) ----
    if (lane < 16) {
        int t   = tbase + lane;
        bool ok = (t < n_tokens);
        int tgt = ok ? target_ids[t] : 0;
        tgt = tgt < 0 ? 0 : (tgt > vocab - 1 ? vocab - 1 : tgt);
        #pragma unroll
        for (int l = 0; l < MAXPATH; ++l) {
            int nd = path_nodes[tgt * MAXPATH + l];
            s_node[lane * MAXPATH + l] = nd < 0 ? 0 : nd;
            s_t[lane * MAXPATH + l]    = path_tgt[tgt * MAXPATH + l];
            // zero mask for out-of-range tail tokens so they contribute nothing
            s_m[lane * MAXPATH + l]    = ok ? path_msk[tgt * MAXPATH + l] : 0.0f;
        }
    }

    // ---- Stage hidden tile as f16 in the contiguous-K B-operand layout this kernel uses ----
    // B is 32x16 (KxN): lane group hi=(lane>>4) supplies K = hi*16..hi*16+15
    // for column n = lane&15, as 16 contiguous halves.
    {
        const int n  = lane & 15;
        const int kh = (lane >> 4) * 16;
        int t = tbase + n;
        if (t >= n_tokens) t = n_tokens - 1;   // clamped; masked out above
        const float* hp0 = hidden + (size_t)t * DIM;
        for (int kc = 0; kc < KCHUNKS; ++kc) {
            const float* hp = hp0 + kc * 32 + kh;
            _Float16* dst = &hB[kc * 512 + lane * 16];
            #pragma unroll
            for (int j = 0; j < 16; ++j) dst[j] = (_Float16)hp[j];
        }
    }
    __syncthreads();

    // Diagonal ownership of the 16x16 f32 D tile:
    //   token m in [0,8):  lane m,     element m of v8f
    //   token m in [8,16): lane m+16,  element m-8
    const bool owner   = (lane < 8) || (lane >= 24);
    const int  m_owner = (lane < 8) ? lane : (lane - 16);
    const int  cidx    = (lane < 8) ? lane : (lane - 24);

    // A-operand addressing: A is 16x32 (MxK): lane hi*16+m supplies
// halves 0..15 = K hi*16 .. hi*16+15 (contiguous, matching B).
    const int ma  = lane & 15;
    const int kha = (lane >> 4) * 16;

    float loss_acc = 0.0f, mask_acc = 0.0f;

    for (int l = 0; l < MAXPATH; ++l) {
        const int row = s_node[ma * MAXPATH + l];
        const float* wrow = node_w + (size_t)row * DIM + kha;

        v8f c = {};
        for (int kc = 0; kc < KCHUNKS; ++kc) {
            const float* wp = wrow + kc * 32;
            v16h a;
            #pragma unroll
            for (int j = 0; j < 8; ++j) {
                a[j]     = (_Float16)wp[j];        // K = kha + j
                a[8 + j] = (_Float16)wp[8 + j];    // K = kha + 8 + j
            }
            v16h b = *(const v16h*)(&hB[kc * 512 + lane * 16]);
            c = __builtin_amdgcn_wmma_f32_16x16x32_f16(
                    false, a, false, b, (short)0, c, false, false);
        }

        // Extract this lane's diagonal element D[m][m]
        float logit = c[0];
        #pragma unroll
        for (int j = 1; j < 8; ++j) logit = (cidx == j) ? c[j] : logit;

        if (owner) {
            float z  = s_t[m_owner * MAXPATH + l];
            float mk = s_m[m_owner * MAXPATH + l];
            float x  = logit;
            // stable BCE-with-logits: max(x,0) - x*z + log1p(exp(-|x|))
            float bce = fmaxf(x, 0.0f) - x * z + log1pf(expf(-fabsf(x)));
            loss_acc += bce * mk;
            mask_acc += mk;
        }
    }

    float validf = (owner && mask_acc > 0.0f) ? 1.0f : 0.0f;
    if (!owner) loss_acc = 0.0f;

    // wave32 tree reduction
    for (int off = 16; off > 0; off >>= 1) {
        loss_acc += __shfl_down(loss_acc, off, 32);
        validf   += __shfl_down(validf,   off, 32);
    }
    if (lane == 0) {
        atomicAdd(&ws[0], loss_acc);
        atomicAdd(&ws[1], validf);
    }
}

__global__ void hs_finalize_kernel(const float* __restrict__ ws, float* __restrict__ out)
{
    float s = ws[0];
    float v = ws[1];
    out[0] = (v > 0.0f) ? (s / fmaxf(v, 1.0f)) : 0.0f;
}

extern "C" void kernel_launch(void* const* d_in, const int* in_sizes, int n_in,
                              void* d_out, int out_size, void* d_ws, size_t ws_size,
                              hipStream_t stream)
{
    const float* hidden     = (const float*)d_in[0];
    const int*   target_ids = (const int*)  d_in[1];
    const float* node_w     = (const float*)d_in[2];
    const int*   path_nodes = (const int*)  d_in[3];
    const float* path_tgt   = (const float*)d_in[4];
    const float* path_msk   = (const float*)d_in[5];

    const int n_tokens = in_sizes[1];                 // B*S = 8192
    const int vocab    = in_sizes[3] / MAXPATH;       // 50257

    hipMemsetAsync(d_ws, 0, 2 * sizeof(float), stream);

    const int nblocks = (n_tokens + 15) / 16;
    hs_path_logits_kernel<<<nblocks, 32, 0, stream>>>(
        hidden, target_ids, node_w, path_nodes, path_tgt, path_msk,
        (float*)d_ws, n_tokens, vocab);

    hs_finalize_kernel<<<1, 1, 0, stream>>>((const float*)d_ws, (float*)d_out);
}
